// SelfAttention_88897233093041
// MI455X (gfx1250) — matrix operand order repaired and checked by reading, not compiled, not run on hardware
//
#include <hip/hip_runtime.h>

typedef __bf16 bf16;
typedef float v8f   __attribute__((ext_vector_type(8)));
typedef bf16  v16bf __attribute__((ext_vector_type(16)));
typedef unsigned int u32x4 __attribute__((ext_vector_type(4)));

union Frag16 {
  v16bf v;
  u32x4 q[2];
};

static __device__ inline v8f vzero8() {
  v8f z;
#pragma unroll
  for (int i = 0; i < 8; i++) z[i] = 0.f;
  return z;
}

static __device__ inline v8f wmma_bf16(const v16bf a, const v16bf b, const v8f c) {
  // emits v_wmma_f32_16x16x32_bf16
  return __builtin_amdgcn_wmma_f32_16x16x32_bf16(false, a, false, b, (short)0, c, false, false);
}

// ---------------------------------------------------------------------------
// Async global->LDS copy (CDNA5 GLOBAL_LOAD_ASYNC_TO_LDS_B128, ASYNCcnt),
// with a synchronous VGPR-roundtrip fallback if the builtin is absent.
// ---------------------------------------------------------------------------
#if defined(__has_builtin)
#if __has_builtin(__builtin_amdgcn_global_load_async_to_lds_b128)
#define HAVE_ASYNC_COPY 1
#endif
#endif

static __device__ inline void copy16_g2l(const void* g, void* l) {
#if defined(HAVE_ASYNC_COPY)
  typedef int v4i __attribute__((__vector_size__(16)));
  typedef __attribute__((address_space(1))) v4i* gp_v4i;
  typedef __attribute__((address_space(3))) v4i* lp_v4i;
  // generic LDS address low 32 bits == LDS byte offset (ISA 10.2 aperture rules)
  __builtin_amdgcn_global_load_async_to_lds_b128(
      (gp_v4i)(uintptr_t)g, (lp_v4i)(unsigned)(uintptr_t)l, 0, 0);
#else
  *(u32x4*)l = *(const u32x4*)g;
#endif
}
static __device__ inline void copy_wait() {
#if defined(HAVE_ASYNC_COPY)
  asm volatile("s_wait_asynccnt 0x0" ::: "memory");
#endif
}

// ---------------------------------------------------------------------------
// 16-lane row max via DPP (xor1/2/4 via DPP8, xor8 via DPP16 ROW_ROR:8)
// ---------------------------------------------------------------------------
#define DPP8_SEL(a, b, c, d, e, f, g, h)                                   \
  ((a) | ((b) << 3) | ((c) << 6) | ((d) << 9) | ((e) << 12) | ((f) << 15) | \
   ((g) << 18) | ((h) << 21))

#if defined(__has_builtin)
#if __has_builtin(__builtin_amdgcn_mov_dpp8) && __has_builtin(__builtin_amdgcn_update_dpp)
#define HAVE_DPP 1
#endif
#endif

static __device__ inline float row_max16(float x) {
#if defined(HAVE_DPP)
  x = fmaxf(x, __int_as_float(__builtin_amdgcn_mov_dpp8(
                   __float_as_int(x), DPP8_SEL(1, 0, 3, 2, 5, 4, 7, 6))));
  x = fmaxf(x, __int_as_float(__builtin_amdgcn_mov_dpp8(
                   __float_as_int(x), DPP8_SEL(2, 3, 0, 1, 6, 7, 4, 5))));
  x = fmaxf(x, __int_as_float(__builtin_amdgcn_mov_dpp8(
                   __float_as_int(x), DPP8_SEL(4, 5, 6, 7, 0, 1, 2, 3))));
  // ROW_ROR:8 within the 16-lane row == xor 8 for a full-row reduction
  x = fmaxf(x, __int_as_float(__builtin_amdgcn_update_dpp(
                   0, __float_as_int(x), 0x128, 0xF, 0xF, true)));
#else
#pragma unroll
  for (int mask = 8; mask >= 1; mask >>= 1)
    x = fmaxf(x, __shfl_xor(x, mask, 32));
#endif
  return x;
}

// ---------------------------------------------------------------------------
// 1) fp32 -> bf16 elementwise convert
// ---------------------------------------------------------------------------
__global__ void cvt_f32_to_bf16(const float* __restrict__ in, bf16* __restrict__ out, int n) {
  int i = blockIdx.x * blockDim.x + threadIdx.x;
  if (i < n) out[i] = (bf16)in[i];
}

// ---------------------------------------------------------------------------
// 2) fp32 [R,C] -> bf16 [C,R] transpose-convert (32x32 LDS tiles)
// ---------------------------------------------------------------------------
__global__ void transpose_cvt(const float* __restrict__ in, bf16* __restrict__ out,
                              int R, int C) {
  __shared__ float tile[32][33];
  const int tx = threadIdx.x, ty = threadIdx.y;
  const int c0 = blockIdx.x * 32, r0 = blockIdx.y * 32;
#pragma unroll
  for (int j = 0; j < 32; j += 8)
    tile[ty + j][tx] = in[(size_t)(r0 + ty + j) * C + c0 + tx];
  __syncthreads();
#pragma unroll
  for (int j = 0; j < 32; j += 8)
    out[(size_t)(c0 + ty + j) * R + r0 + tx] = (bf16)tile[tx][ty + j];
}

// ---------------------------------------------------------------------------
// 3) Tiled WMMA GEMM: C[M,N] = A[M,K](bf16) * BT[N,K](bf16)^T + bias[N]
//    BM=BN=128, BK=32; 8 waves = 2(M) x 4(N), each 64x32 via 4x2 WMMA tiles.
//    Double-buffered async global->LDS staging overlapped with WMMA.
// ---------------------------------------------------------------------------
template <bool OUT_BF16>
__global__ __launch_bounds__(256) void gemm_bf16_wmma(
    const bf16* __restrict__ A, const bf16* __restrict__ BT,
    const float* __restrict__ bias, void* __restrict__ Cv,
    int M, int N, int K) {
  __shared__ alignas(16) bf16 sA[2][128 * 32];
  __shared__ alignas(16) bf16 sB[2][128 * 32];

  const int tid  = threadIdx.x;
  const int w    = tid >> 5;
  const int lane = tid & 31;
  const int lh   = lane >> 4;
  const int ln   = lane & 15;
  const int waveM = w >> 2;
  const int waveN = w & 3;
  const int rowBase = blockIdx.x * 128;
  const int colBase = blockIdx.y * 128;

  const int ldRow  = tid >> 1;        // 0..127
  const int ldHalf = (tid & 1) * 16;  // 16 bf16 per half

  const bf16* agBase = A  + (size_t)(rowBase + ldRow) * K + ldHalf;
  const bf16* bgBase = BT + (size_t)(colBase + ldRow) * K + ldHalf;

  v8f acc[4][2];
#pragma unroll
  for (int i = 0; i < 4; i++)
#pragma unroll
    for (int j = 0; j < 2; j++) acc[i][j] = vzero8();

  // prologue: stage k0=0 into buffer 0
  copy16_g2l(agBase,     sA[0] + ldRow * 32 + ldHalf);
  copy16_g2l(agBase + 8, sA[0] + ldRow * 32 + ldHalf + 8);
  copy16_g2l(bgBase,     sB[0] + ldRow * 32 + ldHalf);
  copy16_g2l(bgBase + 8, sB[0] + ldRow * 32 + ldHalf + 8);

  int buf = 0;
  for (int k0 = 0; k0 < K; k0 += 32, buf ^= 1) {
    copy_wait();      // only chunk-k0 copies outstanding here
    __syncthreads();  // chunk k0 visible; previous compute done -> buf^1 free
    if (k0 + 32 < K) {
      const bf16* ag = agBase + k0 + 32;
      const bf16* bg = bgBase + k0 + 32;
      bf16* dA = sA[buf ^ 1] + ldRow * 32 + ldHalf;
      bf16* dB = sB[buf ^ 1] + ldRow * 32 + ldHalf;
      copy16_g2l(ag,     dA);
      copy16_g2l(ag + 8, dA + 8);
      copy16_g2l(bg,     dB);
      copy16_g2l(bg + 8, dB + 8);
    }

    Frag16 fa[4], fb[2];
#pragma unroll
    for (int i = 0; i < 4; i++) {
      // A layout: lane=row, K = {8*lh..} U {16+8*lh..}
      const bf16* s = sA[buf] + (waveM * 64 + i * 16 + ln) * 32;
      fa[i].q[0] = *(const u32x4*)(s + lh * 8);
      fa[i].q[1] = *(const u32x4*)(s + 16 + lh * 8);
    }
#pragma unroll
    for (int j = 0; j < 2; j++) {
      // B layout: lane=col, K = {8*lh..} U {16+8*lh..}
      const bf16* s = sB[buf] + (waveN * 32 + j * 16 + ln) * 32;
      fb[j].q[0] = *(const u32x4*)(s + lh * 8);
      fb[j].q[1] = *(const u32x4*)(s + 16 + lh * 8);
    }
#pragma unroll
    for (int i = 0; i < 4; i++)
#pragma unroll
      for (int j = 0; j < 2; j++)
        acc[i][j] = wmma_bf16(fa[i].v, fb[j].v, acc[i][j]);
  }

  // Epilogue: D layout -> col = lane%16, row = vgpr + 8*lh
#pragma unroll
  for (int i = 0; i < 4; i++) {
#pragma unroll
    for (int j = 0; j < 2; j++) {
      const int col = colBase + waveN * 32 + j * 16 + ln;
      const float bc = bias[col];
      const int rbase = rowBase + waveM * 64 + i * 16 + lh * 8;
#pragma unroll
      for (int vi = 0; vi < 8; vi++) {
        const float val = acc[i][j][vi] + bc;
        const size_t idx = (size_t)(rbase + vi) * N + col;
        if (OUT_BF16) ((bf16*)Cv)[idx] = (bf16)val;
        else          ((float*)Cv)[idx] = val;
      }
    }
  }
}

// ---------------------------------------------------------------------------
// 4) RMSNorm q/k + scatter: qkvb [B,N,3,H,hd] bf16 ->
//    qn/kn [B,H,N,hd] bf16 (normed+scaled), vT [B,H,hd,N] bf16
// ---------------------------------------------------------------------------
__global__ void rmsnorm_qkv(const bf16* __restrict__ qkvb,
                            const float* __restrict__ qs, const float* __restrict__ ks,
                            bf16* __restrict__ qn, bf16* __restrict__ kn,
                            bf16* __restrict__ vT) {
  const int idx = blockIdx.x * blockDim.x + threadIdx.x;  // B*H*N = 65536
  const int n = idx & 1023;
  const int h = (idx >> 10) & 15;
  const int b = idx >> 14;
  const int bh = (b << 4) | h;
  const bf16* base = qkvb + (size_t)((b << 10) | n) * 3072 + h * 64;

  float buf[64];
  float ss = 0.f;
#pragma unroll
  for (int d = 0; d < 64; d++) { buf[d] = (float)base[d]; ss += buf[d] * buf[d]; }
  float inv = 1.f / (sqrtf(ss * (1.f / 64.f)) + 1e-8f);
  bf16* qo = qn + (size_t)((bh << 10) | n) * 64;
#pragma unroll
  for (int d = 0; d < 64; d++) qo[d] = (bf16)(qs[d] * buf[d] * inv);

  ss = 0.f;
#pragma unroll
  for (int d = 0; d < 64; d++) { buf[d] = (float)base[1024 + d]; ss += buf[d] * buf[d]; }
  inv = 1.f / (sqrtf(ss * (1.f / 64.f)) + 1e-8f);
  bf16* ko = kn + (size_t)((bh << 10) | n) * 64;
#pragma unroll
  for (int d = 0; d < 64; d++) ko[d] = (bf16)(ks[d] * buf[d] * inv);

  bf16* vo = vT + ((size_t)bh << 16) + n;
#pragma unroll
  for (int d = 0; d < 64; d++) vo[(size_t)d << 10] = base[2048 + d];
}

// ---------------------------------------------------------------------------
// 5) Flash attention, per (b,h): 8 waves x 16 query rows per block.
//    K/V chunk staging double-buffered in LDS (async copy overlapped with
//    WMMA); row-sum via ones-WMMA; row-max via DPP; P re-laid-out D->A
//    through wave-private LDS.
// ---------------------------------------------------------------------------
__global__ __launch_bounds__(256) void attn_fwd(
    const bf16* __restrict__ qn, const bf16* __restrict__ kn,
    const bf16* __restrict__ vT, const float* __restrict__ bias,
    bf16* __restrict__ ao) {
  __shared__ alignas(16) bf16 sK[2][32 * 64];   // [key][d]   2x4KB
  __shared__ alignas(16) bf16 sV[2][64 * 32];   // [d][key]   2x4KB
  __shared__ alignas(16) bf16 sP[8 * 16 * 32];  // per-wave P 8KB

  const int bh   = blockIdx.x;  // 0..63
  const int tid  = threadIdx.x;
  const int w    = tid >> 5;
  const int lane = tid & 31;
  const int lh   = lane >> 4;
  const int ln   = lane & 15;
  const int q0   = blockIdx.y * 128 + w * 16;

  bf16* pw = sP + w * (16 * 32);

  // staging assignments (1 x b128 per thread per tile)
  const int kRow = tid >> 3, kCol = (tid & 7) * 8;  // sK: 32x64
  const int vRow = tid >> 2, vCol = (tid & 3) * 8;  // sV: 64x32

  const bf16* kbase = kn + (((size_t)bh << 10) * 64);
  const bf16* vbase = vT + ((size_t)bh << 16);
  const bf16* kg = kbase + (size_t)kRow * 64 + kCol;   // + j*64
  const bf16* vg = vbase + (size_t)vRow * 1024 + vCol; // + j

  // q A-fragments for d-chunks [0,32) and [32,64)
  const bf16* qsrc = qn + ((size_t)(bh << 10) + q0 + ln) * 64;
  Frag16 aq[2];
  aq[0].q[0] = *(const u32x4*)(qsrc + lh * 8);
  aq[0].q[1] = *(const u32x4*)(qsrc + 16 + lh * 8);
  aq[1].q[0] = *(const u32x4*)(qsrc + 32 + lh * 8);
  aq[1].q[1] = *(const u32x4*)(qsrc + 48 + lh * 8);

  Frag16 fOnes;
#pragma unroll
  for (int i = 0; i < 16; i++) fOnes.v[i] = (bf16)1.0f;

  v8f O[4];
#pragma unroll
  for (int t = 0; t < 4; t++) O[t] = vzero8();
  v8f lacc = vzero8();
  float m[8];
#pragma unroll
  for (int vi = 0; vi < 8; vi++) m[vi] = -3.0e38f;

  const float scale = 0.125f;  // hd^-0.5
  const float* bptr = bias + ((size_t)bh * 1024 + q0 + lh * 8) * 1024 + ln;

  // prologue: stage chunk j=0 into buffer 0
  copy16_g2l(kg, sK[0] + kRow * 64 + kCol);
  copy16_g2l(vg, sV[0] + vRow * 32 + vCol);

  int buf = 0;
  for (int j = 0; j < 1024; j += 32, buf ^= 1) {
    copy_wait();      // chunk-j copies done (only ones outstanding)
    __syncthreads();  // chunk j visible; compute(j-32) done -> buf^1 free
    if (j + 32 < 1024) {
      copy16_g2l(kg + (size_t)(j + 32) * 64, sK[buf ^ 1] + kRow * 64 + kCol);
      copy16_g2l(vg + (j + 32),              sV[buf ^ 1] + vRow * 32 + vCol);
    }

    // S = q @ k^T for two 16-key tiles; K=hd=64 via two WMMA each
    v8f s0 = vzero8(), s1 = vzero8();
    {
      const bf16* kp0 = sK[buf] + (ln)*64 + lh * 8;
      const bf16* kp1 = sK[buf] + (16 + ln) * 64 + lh * 8;
      Frag16 kf;
      kf.q[0] = *(const u32x4*)(kp0);      kf.q[1] = *(const u32x4*)(kp0 + 16);
      s0 = wmma_bf16(aq[0].v, kf.v, s0);
      kf.q[0] = *(const u32x4*)(kp0 + 32); kf.q[1] = *(const u32x4*)(kp0 + 48);
      s0 = wmma_bf16(aq[1].v, kf.v, s0);
      kf.q[0] = *(const u32x4*)(kp1);      kf.q[1] = *(const u32x4*)(kp1 + 16);
      s1 = wmma_bf16(aq[0].v, kf.v, s1);
      kf.q[0] = *(const u32x4*)(kp1 + 32); kf.q[1] = *(const u32x4*)(kp1 + 48);
      s1 = wmma_bf16(aq[1].v, kf.v, s1);
    }
    // scale + bias (row = q0 + vi + 8*lh, col = j(+16) + ln)
#pragma unroll
    for (int vi = 0; vi < 8; vi++) {
      s0[vi] = s0[vi] * scale + bptr[(size_t)vi * 1024 + j];
      s1[vi] = s1[vi] * scale + bptr[(size_t)vi * 1024 + j + 16];
    }
    // online softmax: row max via DPP; update m, rescale O and l
#pragma unroll
    for (int vi = 0; vi < 8; vi++) {
      const float cm = row_max16(fmaxf(s0[vi], s1[vi]));
      const float mn = fmaxf(m[vi], cm);
      const float al = __expf(m[vi] - mn);
      m[vi] = mn;
      lacc[vi] *= al;
#pragma unroll
      for (int t = 0; t < 4; t++) O[t][vi] *= al;
      s0[vi] = __expf(s0[vi] - mn);
      s1[vi] = __expf(s1[vi] - mn);
    }
    // P (D-layout) -> wave-private LDS row-major [16][32]
#pragma unroll
    for (int vi = 0; vi < 8; vi++) {
      const int r = vi + lh * 8;
      pw[r * 32 + ln]      = (bf16)s0[vi];
      pw[r * 32 + 16 + ln] = (bf16)s1[vi];
    }
    // reload as A-fragment (DS in-order within wave)
    Frag16 pf;
    const bf16* ps = pw + ln * 32;
    pf.q[0] = *(const u32x4*)(ps + lh * 8);
    pf.q[1] = *(const u32x4*)(ps + 16 + lh * 8);

    // row sums: l += P @ ones  (D rows == per-lane l ownership)
    lacc = wmma_bf16(pf.v, fOnes.v, lacc);

    // O += P @ V
#pragma unroll
    for (int t = 0; t < 4; t++) {
      const bf16* vs = sV[buf] + (t * 16 + ln) * 32 + lh * 8;
      Frag16 vf;
      vf.q[0] = *(const u32x4*)(vs);
      vf.q[1] = *(const u32x4*)(vs + 16);
      O[t] = wmma_bf16(pf.v, vf.v, O[t]);
    }
  }

  // normalize and write bf16 [B,N,C] (head-interleaved) for proj GEMM
  const int b = bh >> 4, h = bh & 15;
#pragma unroll
  for (int vi = 0; vi < 8; vi++) {
    const float il = 1.f / lacc[vi];
    const int row = q0 + vi + lh * 8;
#pragma unroll
    for (int t = 0; t < 4; t++) {
      const int col = h * 64 + t * 16 + ln;
      ao[(size_t)((b << 10) + row) * 1024 + col] = (bf16)(O[t][vi] * il);
    }
  }
}

// ---------------------------------------------------------------------------
// Launch
// ---------------------------------------------------------------------------
extern "C" void kernel_launch(void* const* d_in, const int* in_sizes, int n_in,
                              void* d_out, int out_size, void* d_ws, size_t ws_size,
                              hipStream_t stream) {
  const float* x       = (const float*)d_in[0];
  const float* bias    = (const float*)d_in[1];
  const float* W_qkv   = (const float*)d_in[2];
  const float* b_qkv   = (const float*)d_in[3];
  const float* q_scale = (const float*)d_in[4];
  const float* k_scale = (const float*)d_in[5];
  const float* W_proj  = (const float*)d_in[6];
  const float* b_proj  = (const float*)d_in[7];
  float* out = (float*)d_out;

  char* p = (char*)d_ws;
  auto take = [&](size_t bytes) {
    char* r = p;
    p += (bytes + 255) & ~(size_t)255;
    return r;
  };
  bf16* xb     = (bf16*)take((size_t)4096 * 1024 * 2);
  bf16* wqkvT  = (bf16*)take((size_t)3072 * 1024 * 2);
  bf16* wprojT = (bf16*)take((size_t)1024 * 1024 * 2);
  bf16* qkvb   = (bf16*)take((size_t)4096 * 3072 * 2);
  bf16* qnb    = (bf16*)take((size_t)64 * 1024 * 64 * 2);
  bf16* knb    = (bf16*)take((size_t)64 * 1024 * 64 * 2);
  bf16* vTb    = (bf16*)take((size_t)64 * 64 * 1024 * 2);
  bf16* aob    = (bf16*)take((size_t)4096 * 1024 * 2);

  cvt_f32_to_bf16<<<(4096 * 1024) / 256, 256, 0, stream>>>(x, xb, 4096 * 1024);
  transpose_cvt<<<dim3(3072 / 32, 1024 / 32), dim3(32, 8), 0, stream>>>(W_qkv, wqkvT, 1024, 3072);
  transpose_cvt<<<dim3(1024 / 32, 1024 / 32), dim3(32, 8), 0, stream>>>(W_proj, wprojT, 1024, 1024);

  gemm_bf16_wmma<true><<<dim3(4096 / 128, 3072 / 128), 256, 0, stream>>>(
      xb, wqkvT, b_qkv, qkvb, 4096, 3072, 1024);

  rmsnorm_qkv<<<65536 / 256, 256, 0, stream>>>(qkvb, q_scale, k_scale, qnb, knb, vTb);

  attn_fwd<<<dim3(64, 8), 256, 0, stream>>>(qnb, knb, vTb, bias, aob);

  gemm_bf16_wmma<false><<<dim3(4096 / 128, 1024 / 128), 256, 0, stream>>>(
      aob, wprojT, b_proj, out, 4096, 1024, 1024);
}
